// Gate_52269751992785
// MI455X (gfx1250) — matrix operand order repaired and checked by reading, not compiled, not run on hardware
//
#include <hip/hip_runtime.h>

typedef __attribute__((ext_vector_type(16))) __bf16 v16bf;
typedef __attribute__((ext_vector_type(8)))  float  v8f;
typedef __attribute__((ext_vector_type(4)))  int    v4i;

typedef __attribute__((address_space(1))) v4i* as1_v4i_p;
typedef __attribute__((address_space(3))) v4i* as3_v4i_p;

#define DIM   8192
#define BATCH 256
#define BN    256
#define NBUF  3

struct U32x8 { uint4 a, b; };

__device__ __forceinline__ unsigned int pack_bf16(float lo, float hi) {
    unsigned int ulo = __builtin_bit_cast(unsigned int, lo) + 0x8000u;
    unsigned int uhi = __builtin_bit_cast(unsigned int, hi) + 0x8000u;
    return __builtin_amdgcn_perm(uhi, ulo, 0x07060302u);
}

__device__ __forceinline__ uint2 cvt4(float4 v) {
    uint2 r;
    r.x = pack_bf16(v.x, v.y);
    r.y = pack_bf16(v.z, v.w);
    return r;
}

__device__ __forceinline__ v16bf ld_frag_bf16(const unsigned short* p0, const unsigned short* p1) {
    U32x8 t;
    t.a = *(const uint4*)p0;
    t.b = *(const uint4*)p1;
    return __builtin_bit_cast(v16bf, t);
}

__device__ __forceinline__ v16bf ld_frag_f32(const float* p0, const float* p1) {
    float4 a0 = ((const float4*)p0)[0], a1 = ((const float4*)p0)[1];
    float4 b0 = ((const float4*)p1)[0], b1 = ((const float4*)p1)[1];
    U32x8 t;
    t.a.x = pack_bf16(a0.x, a0.y); t.a.y = pack_bf16(a0.z, a0.w);
    t.a.z = pack_bf16(a1.x, a1.y); t.a.w = pack_bf16(a1.z, a1.w);
    t.b.x = pack_bf16(b0.x, b0.y); t.b.y = pack_bf16(b0.z, b0.w);
    t.b.z = pack_bf16(b1.x, b1.y); t.b.w = pack_bf16(b1.z, b1.w);
    return __builtin_bit_cast(v16bf, t);
}

__device__ __forceinline__ void async_cp16(const void* g, void* l) {
    __builtin_amdgcn_global_load_async_to_lds_b128((as1_v4i_p)g, (as3_v4i_p)l, 0, 0);
}

__global__ __launch_bounds__(256)
void cvt_x_kernel(const float* __restrict__ x, unsigned short* __restrict__ xb) {
    const int idx = blockIdx.x * 256 + threadIdx.x;
    ((uint2*)xb)[idx] = cvt4(((const float4*)x)[idx]);
}

// ====================== main path: x preconverted to bf16 ======================
// BM=32, KT=128 -> 256 blocks, 2x80KB async in flight per block (~40MB chip-wide)
__global__ __launch_bounds__(256)
void gemm_bf16x(const unsigned short* __restrict__ xb,
                const float* __restrict__ U,
                float* __restrict__ out) {
    constexpr int BMk = 32, KTk = 128, NTk = DIM / KTk;   // 64 stages
    constexpr int NPS = 16;   // = #B async ops/stage; A's 4 issued first

    __shared__ float          Ashf[NBUF][BMk * KTk];      // 3 x 16 KB (async fp32 U)
    __shared__ unsigned short Ab[2][BMk * KTk];           // 2 x  8 KB (bf16 converted)
    __shared__ unsigned short Bsh[NBUF][BN * KTk];        // 3 x 64 KB (async bf16 x)

    const int tid   = threadIdx.x;
    const int lane  = tid & 31;
    const int wave  = tid >> 5;
    const int iBase = blockIdx.x * BMk;
    const int waveN = wave * 32;          // 8 waves tile the 256-batch

    const int fRow = lane & 15;
    const int aK   = (lane >> 4) * 16;
    const int bK   = (lane >> 4) * 16;

    v8f acc[2][2];
    const v8f vzero = {0.f, 0.f, 0.f, 0.f, 0.f, 0.f, 0.f, 0.f};
#pragma unroll
    for (int mi = 0; mi < 2; ++mi)
#pragma unroll
        for (int ni = 0; ni < 2; ++ni)
            acc[mi][ni] = vzero;

    auto issue_stage = [&](int s) {       // A first, then B (NPS ordering trick)
        const int buf = s % NBUF;
        const int k0  = s * KTk;
#pragma unroll
        for (int c = 0; c < 4; ++c) {     // A: 32x128 fp32, 4 b128/thread
            const int f = c * 256 + tid, row = f >> 5, col = (f & 31) * 4;
            async_cp16(U + (size_t)(iBase + row) * DIM + k0 + col,
                       &Ashf[buf][row * KTk + col]);
        }
#pragma unroll
        for (int c = 0; c < 16; ++c) {    // B: 256x128 bf16, 16 b128/thread
            const int f = c * 256 + tid, row = f >> 4, col = (f & 15) * 8;
            async_cp16(xb + (size_t)row * DIM + k0 + col,
                       &Bsh[buf][row * KTk + col]);
        }
    };

    auto convert_A = [&](int s) {         // Ashf[s%3] fp32 -> Ab[s&1] bf16
        const float*    src = Ashf[s % NBUF];
        unsigned short* dst = Ab[s & 1];
#pragma unroll
        for (int c = 0; c < 4; ++c) {
            const int f = c * 256 + tid, row = f >> 5, col = (f & 31) * 4;
            float4 v = *(const float4*)&src[row * KTk + col];
            *(uint2*)&dst[row * KTk + col] = cvt4(v);
        }
    };

    issue_stage(0);
    issue_stage(1);
    __builtin_amdgcn_s_wait_asynccnt(NPS);   // stage0 landed + stage1's A landed
    convert_A(0);

    for (int t = 0; t < NTk; ++t) {
        __builtin_amdgcn_s_wait_asynccnt(NPS);
        __syncthreads();

        const unsigned short* Aw = Ab[t & 1];
        const unsigned short* Bb = Bsh[t % NBUF];

#pragma unroll
        for (int kk = 0; kk < KTk; kk += 32) {
            v16bf afrag[2], bfrag[2];
#pragma unroll
            for (int mi = 0; mi < 2; ++mi) {
                const unsigned short* p = &Aw[(mi * 16 + fRow) * KTk + kk + aK];
                afrag[mi] = ld_frag_bf16(p, p + 8);
            }
#pragma unroll
            for (int ni = 0; ni < 2; ++ni) {
                const unsigned short* p = &Bb[(waveN + ni * 16 + fRow) * KTk + kk + bK];
                bfrag[ni] = ld_frag_bf16(p, p + 8);
            }
#pragma unroll
            for (int mi = 0; mi < 2; ++mi)
#pragma unroll
                for (int ni = 0; ni < 2; ++ni)
                    acc[mi][ni] = __builtin_amdgcn_wmma_f32_16x16x32_bf16(
                        false, afrag[mi], false, bfrag[ni],
                        (short)0, acc[mi][ni], false, false);
        }

        if (t + 1 < NTk) convert_A(t + 1);
        if (t + 2 < NTk) issue_stage(t + 2);
    }

#pragma unroll
    for (int mi = 0; mi < 2; ++mi) {
#pragma unroll
        for (int ni = 0; ni < 2; ++ni) {
            const int i = iBase + mi * 16 + (lane >> 4) * 8;
            const int b = waveN + ni * 16 + (lane & 15);
            float* o = out + (size_t)b * DIM + i;
            v8f c = acc[mi][ni];
            float4 lo = {c[0], c[1], c[2], c[3]};
            float4 hi = {c[4], c[5], c[6], c[7]};
            *(float4*)(o)     = lo;
            *(float4*)(o + 4) = hi;
        }
    }
}

// ============ fallback: x stays fp32 (ws too small), BM=64 / KT=64 ============
__global__ __launch_bounds__(256)
void gemm_f32x(const float* __restrict__ x32,
               const float* __restrict__ U,
               float* __restrict__ out) {
    constexpr int BMk = 64, KTk = 64, NTk = DIM / KTk;
    constexpr int NPS = 16;   // = #B async ops/stage; A's 4 issued first

    __shared__ float          Ashf[NBUF][BMk * KTk];      // 3 x 16 KB
    __shared__ unsigned short Ab[2][BMk * KTk];           // 2 x  8 KB
    __shared__ float          Bsh[NBUF][BN * KTk];        // 3 x 64 KB (fp32 x)

    const int tid   = threadIdx.x;
    const int lane  = tid & 31;
    const int wave  = tid >> 5;
    const int iBase = blockIdx.x * BMk;

    const int waveM = (wave & 1) * 32;
    const int waveN = (wave >> 1) * 64;

    const int fRow = lane & 15;
    const int aK   = (lane >> 4) * 16;
    const int bK   = (lane >> 4) * 16;

    v8f acc[2][4];
    const v8f vzero = {0.f, 0.f, 0.f, 0.f, 0.f, 0.f, 0.f, 0.f};
#pragma unroll
    for (int mi = 0; mi < 2; ++mi)
#pragma unroll
        for (int ni = 0; ni < 4; ++ni)
            acc[mi][ni] = vzero;

    auto issue_stage = [&](int s) {
        const int buf = s % NBUF;
        const int k0  = s * KTk;
#pragma unroll
        for (int c = 0; c < 4; ++c) {     // A: 64x64 fp32
            const int f = c * 256 + tid, row = f >> 4, col = (f & 15) * 4;
            async_cp16(U + (size_t)(iBase + row) * DIM + k0 + col,
                       &Ashf[buf][row * KTk + col]);
        }
#pragma unroll
        for (int c = 0; c < 16; ++c) {    // B: 256x64 fp32
            const int f = c * 256 + tid, row = f >> 4, col = (f & 15) * 4;
            async_cp16(x32 + (size_t)row * DIM + k0 + col,
                       &Bsh[buf][row * KTk + col]);
        }
    };

    auto convert_A = [&](int s) {
        const float*    src = Ashf[s % NBUF];
        unsigned short* dst = Ab[s & 1];
#pragma unroll
        for (int c = 0; c < 4; ++c) {
            const int f = c * 256 + tid, row = f >> 4, col = (f & 15) * 4;
            float4 v = *(const float4*)&src[row * KTk + col];
            *(uint2*)&dst[row * KTk + col] = cvt4(v);
        }
    };

    issue_stage(0);
    issue_stage(1);
    __builtin_amdgcn_s_wait_asynccnt(NPS);
    convert_A(0);

    for (int t = 0; t < NTk; ++t) {
        __builtin_amdgcn_s_wait_asynccnt(NPS);
        __syncthreads();

        const unsigned short* Aw = Ab[t & 1];
        const float*          Bf = Bsh[t % NBUF];

#pragma unroll
        for (int kk = 0; kk < KTk; kk += 32) {
            v16bf afrag[2], bfrag[4];
#pragma unroll
            for (int mi = 0; mi < 2; ++mi) {
                const unsigned short* p = &Aw[(waveM + mi * 16 + fRow) * KTk + kk + aK];
                afrag[mi] = ld_frag_bf16(p, p + 8);
            }
#pragma unroll
            for (int ni = 0; ni < 4; ++ni) {
                const float* p = &Bf[(waveN + ni * 16 + fRow) * KTk + kk + bK];
                bfrag[ni] = ld_frag_f32(p, p + 8);
            }
#pragma unroll
            for (int mi = 0; mi < 2; ++mi)
#pragma unroll
                for (int ni = 0; ni < 4; ++ni)
                    acc[mi][ni] = __builtin_amdgcn_wmma_f32_16x16x32_bf16(
                        false, afrag[mi], false, bfrag[ni],
                        (short)0, acc[mi][ni], false, false);
        }

        if (t + 1 < NTk) convert_A(t + 1);
        if (t + 2 < NTk) issue_stage(t + 2);
    }

#pragma unroll
    for (int mi = 0; mi < 2; ++mi) {
#pragma unroll
        for (int ni = 0; ni < 4; ++ni) {
            const int i = iBase + waveM + mi * 16 + (lane >> 4) * 8;
            const int b = waveN + ni * 16 + (lane & 15);
            float* o = out + (size_t)b * DIM + i;
            v8f c = acc[mi][ni];
            float4 lo = {c[0], c[1], c[2], c[3]};
            float4 hi = {c[4], c[5], c[6], c[7]};
            *(float4*)(o)     = lo;
            *(float4*)(o + 4) = hi;
        }
    }
}

extern "C" void kernel_launch(void* const* d_in, const int* in_sizes, int n_in,
                              void* d_out, int out_size, void* d_ws, size_t ws_size,
                              hipStream_t stream) {
    const float* x = (const float*)d_in[0];   // [256, 8192]
    const float* U = (const float*)d_in[1];   // [8192, 8192]
    float* out     = (float*)d_out;           // [256, 8192, 1]

    const size_t xb_bytes = (size_t)BATCH * DIM * sizeof(unsigned short);  // 4 MB

    if (ws_size >= xb_bytes) {
        unsigned short* xb = (unsigned short*)d_ws;
        cvt_x_kernel<<<dim3(BATCH * DIM / 4 / 256), dim3(256), 0, stream>>>(x, xb);
        gemm_bf16x<<<dim3(DIM / 32), dim3(256), 0, stream>>>(xb, U, out);   // 256 blocks
    } else {
        gemm_f32x<<<dim3(DIM / 64), dim3(256), 0, stream>>>(x, U, out);     // 128 blocks
    }
}
